// ScaledAttentionLayer_72009421684774
// MI455X (gfx1250) — matrix operand order repaired and checked by reading, not compiled, not run on hardware
//
#include <hip/hip_runtime.h>
#include <hip/hip_bf16.h>
#include <math.h>

// ---------------------------------------------------------------------------
// ScaledAttentionLayer on MI455X (gfx1250, wave32, WMMA).
// B=8, CQ=CK=2048, D=512.  All matrix math via v_wmma_f32_16x16x32_bf16
// (f32 accumulate).  Flash-attention online softmax: scores never touch HBM.
// ---------------------------------------------------------------------------

typedef __bf16 bf16_t;
typedef __attribute__((ext_vector_type(16))) __bf16        v16bf;
typedef __attribute__((ext_vector_type(8)))  float         v8f;
typedef __attribute__((ext_vector_type(4)))  unsigned int  v4u;

union F16x16 { v16bf bf; v4u u[2]; };   // one 16x32 bf16 WMMA operand (32 B/lane)

#define B_N   8
#define CQ_N  2048
#define CK_N  2048
#define D_N   512
#define NQ    (B_N * CQ_N)   /* 16384 flattened query rows  */
#define NK    (B_N * CK_N)   /* 16384 flattened context rows */
#define CHUNK 256            /* KV tile width per attention iteration */

__device__ __forceinline__ v8f wmma_bf16(const F16x16& a, const F16x16& b, v8f c) {
  // D = A(16x32 bf16) * B(32x16 bf16) + C(16x16 f32)
  return __builtin_amdgcn_wmma_f32_16x16x32_bf16(
      /*neg_a=*/false, a.bf, /*neg_b=*/false, b.bf,
      /*c_mod=*/(short)0, c, /*reuse_a=*/false, /*reuse_b=*/false);
}

// ---------------------------------------------------------------------------
// Phase 0: f32 -> bf16 conversion (vectorized 4-wide; counts are /4 exact)
// ---------------------------------------------------------------------------
__global__ void cvt_f32_to_bf16(const float* __restrict__ src,
                                bf16_t* __restrict__ dst, int n4) {
  int i = blockIdx.x * blockDim.x + threadIdx.x;
  if (i >= n4) return;
  float4 f = ((const float4*)src)[i];
  bf16_t o[4] = { (bf16_t)f.x, (bf16_t)f.y, (bf16_t)f.z, (bf16_t)f.w };
  *(uint2*)(dst + (size_t)i * 4) = *(uint2*)o;
}

// ---------------------------------------------------------------------------
// Phase 1: Y = X * W^T + bias   (bf16 in, bf16 out, optional scale/transpose)
//   X: [rows][512] bf16, W: [512][512] bf16 row-major ([n][k]).
//   Each wave owns a 16x32 output tile: one A-frag feeds TWO B-frags per
//   K-step (2 WMMAs) -> halves A traffic per WMMA vs 16x16 tiling.
//   transposeStore=1 writes Y^T per batch: dst[(b*D + n)*CK + ck]  (for V).
// ---------------------------------------------------------------------------
__global__ __launch_bounds__(256)
void gemm_bf16_kernel(const bf16_t* __restrict__ A,
                      const bf16_t* __restrict__ W,
                      const float*  __restrict__ bias,
                      bf16_t* __restrict__ out,
                      int rows, float outScale, int transposeStore) {
  const int lane = threadIdx.x & 31;
  const int wave = threadIdx.x >> 5;
  const int n16  = lane & 15;
  const int half = lane >> 4;

  const int nPairs = D_N / 32;                    // 16 column-pair tiles
  long tileId  = (long)blockIdx.x * 8 + wave;
  int  rowTile = (int)(tileId / nPairs);
  int  colPair = (int)(tileId % nPairs);
  int  row0 = rowTile * 16;
  int  col0 = colPair * 32;
  if (row0 >= rows) return;

  const bf16_t* arow  = A + (size_t)(row0 + n16) * D_N;        // A row m
  const bf16_t* wrow0 = W + (size_t)(col0 + n16) * D_N;        // B cols 0..15
  const bf16_t* wrow1 = wrow0 + (size_t)16 * D_N;              // B cols 16..31

  v8f acc0, acc1;
  #pragma unroll
  for (int r = 0; r < 8; ++r) { acc0[r] = 0.f; acc1[r] = 0.f; }

  #pragma unroll 4
  for (int k = 0; k < D_N; k += 32) {
    F16x16 a, b0, b1;
    a.u[0]  = *(const v4u*)(arow + k + half * 16);       // K = k+half*16 ..
    a.u[1]  = *(const v4u*)(arow + k + half * 16 + 8);
    b0.u[0] = *(const v4u*)(wrow0 + k + half * 16);      // K = k+half*16 ..
    b0.u[1] = *(const v4u*)(wrow0 + k + half * 16 + 8);
    b1.u[0] = *(const v4u*)(wrow1 + k + half * 16);
    b1.u[1] = *(const v4u*)(wrow1 + k + half * 16 + 8);
    acc0 = wmma_bf16(a, b0, acc0);
    acc1 = wmma_bf16(a, b1, acc1);
  }

  const float bv0 = bias[col0 + n16];
  const float bv1 = bias[col0 + 16 + n16];
  #pragma unroll
  for (int r = 0; r < 8; ++r) {
    float y0 = (acc0[r] + bv0) * outScale;
    float y1 = (acc1[r] + bv1) * outScale;
    int row = row0 + r + 8 * half;                       // C row = r + 8*half
    if (transposeStore) {
      int bb = row >> 11;                                // row / CK_N (2048)
      int ck = row & (CK_N - 1);
      out[((size_t)bb * D_N + (col0 + n16)) * CK_N + ck]      = (bf16_t)y0;
      out[((size_t)bb * D_N + (col0 + 16 + n16)) * CK_N + ck] = (bf16_t)y1;
    } else {
      out[(size_t)row * D_N + (col0 + n16)]      = (bf16_t)y0;
      out[(size_t)row * D_N + (col0 + 16 + n16)] = (bf16_t)y1;
    }
  }
}

// ---------------------------------------------------------------------------
// Phase 3: same 16x32-per-wave GEMM but f32 output to d_out (Wo projection)
// ---------------------------------------------------------------------------
__global__ __launch_bounds__(256)
void gemm_f32out_kernel(const bf16_t* __restrict__ A,
                        const bf16_t* __restrict__ W,
                        const float*  __restrict__ bias,
                        float* __restrict__ out, int rows) {
  const int lane = threadIdx.x & 31;
  const int wave = threadIdx.x >> 5;
  const int n16  = lane & 15;
  const int half = lane >> 4;

  const int nPairs = D_N / 32;
  long tileId  = (long)blockIdx.x * 8 + wave;
  int  rowTile = (int)(tileId / nPairs);
  int  colPair = (int)(tileId % nPairs);
  int  row0 = rowTile * 16;
  int  col0 = colPair * 32;
  if (row0 >= rows) return;

  const bf16_t* arow  = A + (size_t)(row0 + n16) * D_N;
  const bf16_t* wrow0 = W + (size_t)(col0 + n16) * D_N;
  const bf16_t* wrow1 = wrow0 + (size_t)16 * D_N;

  v8f acc0, acc1;
  #pragma unroll
  for (int r = 0; r < 8; ++r) { acc0[r] = 0.f; acc1[r] = 0.f; }

  #pragma unroll 4
  for (int k = 0; k < D_N; k += 32) {
    F16x16 a, b0, b1;
    a.u[0]  = *(const v4u*)(arow + k + half * 16);
    a.u[1]  = *(const v4u*)(arow + k + half * 16 + 8);
    b0.u[0] = *(const v4u*)(wrow0 + k + half * 16);
    b0.u[1] = *(const v4u*)(wrow0 + k + half * 16 + 8);
    b1.u[0] = *(const v4u*)(wrow1 + k + half * 16);
    b1.u[1] = *(const v4u*)(wrow1 + k + half * 16 + 8);
    acc0 = wmma_bf16(a, b0, acc0);
    acc1 = wmma_bf16(a, b1, acc1);
  }

  const float bv0 = bias[col0 + n16];
  const float bv1 = bias[col0 + 16 + n16];
  #pragma unroll
  for (int r = 0; r < 8; ++r) {
    int row = row0 + r + 8 * half;
    out[(size_t)row * D_N + (col0 + n16)]      = acc0[r] + bv0;
    out[(size_t)row * D_N + (col0 + 16 + n16)] = acc1[r] + bv1;
  }
}

// ---------------------------------------------------------------------------
// Phase 2: flash attention.
//   Block = 256 threads = 8 waves, one 16-row Q tile per block.
//   qb: [NQ][D] bf16 (already scaled by 1/sqrt(D));  kb = raw context bf16
//   vT: [B][D][CK] bf16 (transposed V so P*V B-operand loads are contiguous)
//   Per 256-wide KV chunk: wave w computes scores for cols [w*32, w*32+32),
//   block-wide online softmax (max/sum via LDS), P (bf16) shared in LDS,
//   each wave accumulates its 64-column D slice of P*V (4 f32 C tiles).
//   Next chunk's K/V lines are pulled up via global_prefetch.
// ---------------------------------------------------------------------------
__global__ __launch_bounds__(256)
void flash_attn_kernel(const bf16_t* __restrict__ qb,
                       const bf16_t* __restrict__ kb,
                       const bf16_t* __restrict__ vT,
                       bf16_t* __restrict__ ob) {
  __shared__ bf16_t qs[16 * D_N];       // 16 KB: Q tile
  __shared__ bf16_t Ps[16 * CHUNK];     //  8 KB: probability tile (bf16)
  __shared__ float  wmax[8][16];        // per-wave row maxes
  __shared__ float  wsum[8][16];        // per-wave row sums

  const int lane = threadIdx.x & 31;
  const int wave = threadIdx.x >> 5;
  const int n16  = lane & 15;
  const int half = lane >> 4;

  const int qtile = blockIdx.x;                 // 0 .. NQ/16-1
  const int b     = qtile / (CQ_N / 16);
  const size_t qrow0 = (size_t)qtile * 16;      // flattened row base

  // cooperative Q-tile load: 16*512 bf16 = 1024 x 16B
  {
    const uint4* src = (const uint4*)(qb + qrow0 * D_N);
    uint4* dst = (uint4*)qs;
    for (int i = threadIdx.x; i < 1024; i += 256) dst[i] = src[i];
  }
  __syncthreads();

  float runM[8], runL[8];
  #pragma unroll
  for (int r = 0; r < 8; ++r) { runM[r] = -3.0e38f; runL[r] = 0.f; }
  v8f acc[4];
  #pragma unroll
  for (int t = 0; t < 4; ++t)
    #pragma unroll
    for (int r = 0; r < 8; ++r) acc[t][r] = 0.f;

  const int dbase = wave * 64;                  // this wave's D slice
  const bf16_t* vbase = vT + (size_t)b * D_N * CK_N;
  const bf16_t* vrow[4];
  #pragma unroll
  for (int t = 0; t < 4; ++t)
    vrow[t] = vbase + (size_t)(dbase + t * 16 + n16) * CK_N;

  const bf16_t* kbb   = kb + (size_t)b * CK_N * D_N;
  const bf16_t* qsrow = qs + (size_t)n16 * D_N;

  for (int ck0 = 0; ck0 < CK_N; ck0 += CHUNK) {
    // --- 0) prefetch next chunk's K rows / V lines (global_prefetch_b8) ---
    if (ck0 + CHUNK < CK_N) {
      const bf16_t* nk = kbb + (size_t)(ck0 + CHUNK + wave * 32 + n16) * D_N;
      __builtin_prefetch(nk + half * (D_N / 2), 0, 1);          // both halves
      __builtin_prefetch(nk + 16 * D_N + half * (D_N / 2), 0, 1);
      __builtin_prefetch(vrow[half * 2] + ck0 + CHUNK, 0, 1);
      __builtin_prefetch(vrow[half * 2 + 1] + ck0 + CHUNK, 0, 1);
    }

    // --- 1) this wave's 16x32 score tile: S = q * K^T ---------------------
    const int ckw = ck0 + wave * 32;
    const bf16_t* kp0 = kbb + (size_t)(ckw + n16) * D_N;         // cols 0..15
    const bf16_t* kp1 = kp0 + (size_t)16 * D_N;                  // cols 16..31
    v8f s0, s1;
    #pragma unroll
    for (int r = 0; r < 8; ++r) { s0[r] = 0.f; s1[r] = 0.f; }
    #pragma unroll 4
    for (int k = 0; k < D_N; k += 32) {
      F16x16 a, b0, b1;
      a.u[0]  = *(const v4u*)(qsrow + k + half * 16);      // DS loads
      a.u[1]  = *(const v4u*)(qsrow + k + half * 16 + 8);
      b0.u[0] = *(const v4u*)(kp0 + k + half * 16);
      b0.u[1] = *(const v4u*)(kp0 + k + half * 16 + 8);
      b1.u[0] = *(const v4u*)(kp1 + k + half * 16);
      b1.u[1] = *(const v4u*)(kp1 + k + half * 16 + 8);
      s0 = wmma_bf16(a, b0, s0);
      s1 = wmma_bf16(a, b1, s1);
    }

    // --- 2) per-row local max (reduce across 16 lanes of this half) -------
    #pragma unroll
    for (int r = 0; r < 8; ++r) {
      float mv = fmaxf(s0[r], s1[r]);
      mv = fmaxf(mv, __shfl_xor(mv, 1, 32));
      mv = fmaxf(mv, __shfl_xor(mv, 2, 32));
      mv = fmaxf(mv, __shfl_xor(mv, 4, 32));
      mv = fmaxf(mv, __shfl_xor(mv, 8, 32));
      if (n16 == 0) wmax[wave][r + 8 * half] = mv;
    }
    __syncthreads();

    // --- 3) block row max, online-softmax rescale factors -----------------
    float newM[8], alpha[8];
    #pragma unroll
    for (int r = 0; r < 8; ++r) {
      const int row = r + 8 * half;
      float cm = wmax[0][row];
      #pragma unroll
      for (int w = 1; w < 8; ++w) cm = fmaxf(cm, wmax[w][row]);
      float nm = fmaxf(runM[r], cm);
      alpha[r] = __expf(runM[r] - nm);
      newM[r]  = nm;
      runM[r]  = nm;
    }

    // --- 4) P = exp(S - newM) -> bf16 to LDS, per-wave row sums -----------
    #pragma unroll
    for (int r = 0; r < 8; ++r) {
      float p0 = __expf(s0[r] - newM[r]);
      float p1 = __expf(s1[r] - newM[r]);
      const int row = r + 8 * half;
      Ps[row * CHUNK + wave * 32 + n16]      = (bf16_t)p0;
      Ps[row * CHUNK + wave * 32 + 16 + n16] = (bf16_t)p1;
      float sv = p0 + p1;
      sv += __shfl_xor(sv, 1, 32);
      sv += __shfl_xor(sv, 2, 32);
      sv += __shfl_xor(sv, 4, 32);
      sv += __shfl_xor(sv, 8, 32);
      if (n16 == 0) wsum[wave][row] = sv;
    }
    __syncthreads();

    // --- 5) running sum update + accumulator rescale ----------------------
    #pragma unroll
    for (int r = 0; r < 8; ++r) {
      const int row = r + 8 * half;
      float cs = 0.f;
      #pragma unroll
      for (int w = 0; w < 8; ++w) cs += wsum[w][row];
      runL[r] = runL[r] * alpha[r] + cs;
      #pragma unroll
      for (int t = 0; t < 4; ++t) acc[t][r] *= alpha[r];
    }

    // --- 6) acc += P(16x256) * V(256x64 slice) via WMMA -------------------
    #pragma unroll
    for (int ks = 0; ks < CHUNK / 32; ++ks) {
      F16x16 a;
      const bf16_t* prow = Ps + (size_t)n16 * CHUNK + ks * 32;
      a.u[0] = *(const v4u*)(prow + half * 16);
      a.u[1] = *(const v4u*)(prow + half * 16 + 8);
      #pragma unroll
      for (int t = 0; t < 4; ++t) {
        F16x16 bfr;
        const bf16_t* vp = vrow[t] + ck0 + ks * 32 + half * 16;
        bfr.u[0] = *(const v4u*)(vp);
        bfr.u[1] = *(const v4u*)(vp + 8);
        acc[t] = wmma_bf16(a, bfr, acc[t]);
      }
    }
    __syncthreads();   // protect wmax/wsum/Ps before next chunk
  }

  // --- epilogue: normalize and store pre-Wo output (bf16) -----------------
  #pragma unroll
  for (int r = 0; r < 8; ++r) {
    const float inv = 1.0f / runL[r];
    const size_t row = qrow0 + r + 8 * half;
    #pragma unroll
    for (int t = 0; t < 4; ++t)
      ob[row * D_N + dbase + t * 16 + n16] = (bf16_t)(acc[t][r] * inv);
  }
}

// ---------------------------------------------------------------------------
// Host launcher
// ---------------------------------------------------------------------------
extern "C" void kernel_launch(void* const* d_in, const int* in_sizes, int n_in,
                              void* d_out, int out_size, void* d_ws, size_t ws_size,
                              hipStream_t stream) {
  (void)in_sizes; (void)n_in; (void)out_size; (void)ws_size;
  const float* query   = (const float*)d_in[0];
  const float* context = (const float*)d_in[1];
  const float* Wq      = (const float*)d_in[2];
  const float* bq      = (const float*)d_in[3];
  const float* Wv      = (const float*)d_in[4];
  const float* bv      = (const float*)d_in[5];
  const float* Wo      = (const float*)d_in[6];
  const float* bo      = (const float*)d_in[7];
  float* out = (float*)d_out;

  // Workspace carve-up (bf16 staging buffers, 256B aligned): ~82 MB total.
  char* ws = (char*)d_ws;
  size_t off = 0;
  auto carve = [&](size_t bytes) -> void* {
    void* p = ws + off;
    off += (bytes + 255) & ~(size_t)255;
    return p;
  };
  bf16_t* qxb = (bf16_t*)carve((size_t)NQ * D_N * 2);   // bf16(query)
  bf16_t* cxb = (bf16_t*)carve((size_t)NK * D_N * 2);   // bf16(context) == keys
  bf16_t* qb  = (bf16_t*)carve((size_t)NQ * D_N * 2);   // (query Wq^T + bq)/sqrt(D)
  bf16_t* vTb = (bf16_t*)carve((size_t)NK * D_N * 2);   // V^T  [B][D][CK]
  bf16_t* obb = (bf16_t*)carve((size_t)NQ * D_N * 2);   // attention output pre-Wo
  bf16_t* Wqb = (bf16_t*)carve((size_t)D_N * D_N * 2);
  bf16_t* Wvb = (bf16_t*)carve((size_t)D_N * D_N * 2);
  bf16_t* Wob = (bf16_t*)carve((size_t)D_N * D_N * 2);

  // Phase 0: conversions
  const int n4x = NQ * D_N / 4;     // 2,097,152
  const int n4w = D_N * D_N / 4;    // 65,536
  cvt_f32_to_bf16<<<n4x / 256, 256, 0, stream>>>(query,   qxb, n4x);
  cvt_f32_to_bf16<<<n4x / 256, 256, 0, stream>>>(context, cxb, n4x);
  cvt_f32_to_bf16<<<n4w / 256, 256, 0, stream>>>(Wq, Wqb, n4w);
  cvt_f32_to_bf16<<<n4w / 256, 256, 0, stream>>>(Wv, Wvb, n4w);
  cvt_f32_to_bf16<<<n4w / 256, 256, 0, stream>>>(Wo, Wob, n4w);

  // Phase 1: projections (1/sqrt(512) folded into Q)
  const float qscale = 0.04419417382415922f;
  const int gemmBlocks = (NQ / 16) * (D_N / 32) / 8;    // 2048
  gemm_bf16_kernel<<<gemmBlocks, 256, 0, stream>>>(qxb, Wqb, bq, qb,  NQ, qscale, 0);
  gemm_bf16_kernel<<<gemmBlocks, 256, 0, stream>>>(cxb, Wvb, bv, vTb, NK, 1.0f,   1);

  // Phase 2: flash attention (keys = raw bf16 context, per reference)
  flash_attn_kernel<<<NQ / 16, 256, 0, stream>>>(qb, cxb, vTb, obb);

  // Phase 3: output projection, f32 result
  gemm_f32out_kernel<<<gemmBlocks, 256, 0, stream>>>(obb, Wob, bo, out, NQ);
}
